// Model_19842748908308
// MI455X (gfx1250) — matrix operand order repaired and checked by reading, not compiled, not run on hardware
//
#include <hip/hip_runtime.h>
#include <hip/hip_bf16.h>

#define S_  10
#define T_  256
#define B_  128
#define H_  128
#define G4H 512

typedef __attribute__((ext_vector_type(16))) __bf16          v16bf;
typedef __attribute__((ext_vector_type(16))) unsigned short  v16us;
typedef __attribute__((ext_vector_type(8)))  float           v8f;
typedef __attribute__((ext_vector_type(4)))  unsigned int    v4u;
typedef __attribute__((ext_vector_type(8)))  int             v8i;
typedef __attribute__((ext_vector_type(4)))  int             v4i;

__device__ __forceinline__ unsigned short f2bf(float f) {
  unsigned int u = __builtin_bit_cast(unsigned int, f);
  u += 0x7FFFu + ((u >> 16) & 1u);          // round-to-nearest-even
  return (unsigned short)(u >> 16);
}
__device__ __forceinline__ float sigm(float x) { return 1.0f / (1.0f + expf(-x)); }

__device__ __forceinline__ v8f wmma_bf16(v16bf a, v16bf b, v8f c) {
  return __builtin_amdgcn_wmma_f32_16x16x32_bf16(false, a, false, b, (short)0, c, false, false);
}

// A fragment: 16x32 bf16, source row-major [16][ldk] (m, k). ISA 7.12.2 layout.
__device__ __forceinline__ v16bf load_a(const unsigned short* as, int ldk, int k0, int lane) {
  int m = lane & 15, half = lane >> 4;
  const unsigned short* p = as + m * ldk + k0 + half * 8;
  v16us v;
#pragma unroll
  for (int e = 0; e < 8; e++) { v[e] = p[e]; v[e + 8] = p[16 + e]; }
  return __builtin_bit_cast(v16bf, v);
}

// B fragment: 32x16 bf16, source layout [n][ldk] (k contiguous per n).
__device__ __forceinline__ v16bf load_b(const unsigned short* bs, int ldk, int n0, int k0, int lane) {
  int n = lane & 15, kg = lane >> 4;
  const unsigned short* p = bs + (size_t)(n0 + n) * ldk + k0 + kg * 8;
  v16us v;
#pragma unroll
  for (int e = 0; e < 8; e++) { v[e] = p[e]; v[e + 8] = p[16 + e]; }
  return __builtin_bit_cast(v16bf, v);
}

// C fragment store to f32 [m][ldn]
__device__ __forceinline__ void store_c(float* cs, int ldn, int n0, int lane, v8f acc) {
  int n = n0 + (lane & 15), mh = (lane >> 4) * 8;
#pragma unroll
  for (int r = 0; r < 8; r++) cs[(mh + r) * ldn + n] = acc[r];
}

// TDM: async DMA of a [rows x 128] bf16 tile (row-major, tensor row stride 128 elems)
// from global into LDS with 16B padding after each 256B row (-> 136-elem LDS pitch).
// D# per CDNA5 ISA 8.3/8.4; 2D tensor so groups 2/3 (and extra group) are zero.
__device__ __forceinline__ void tdm_load_tile_bf16_r128(const void* gsrc, void* ldsDst,
                                                        unsigned rows, unsigned tensorRows) {
  unsigned long long ga = (unsigned long long)(uintptr_t)gsrc;
  unsigned lds = (unsigned)(uintptr_t)ldsDst;
  v4u g0;
  g0[0] = 1u;                                              // count=1, user mode
  g0[1] = lds;                                             // lds_addr
  g0[2] = (unsigned)ga;                                    // global_addr[31:0]
  g0[3] = (unsigned)((ga >> 32) & 0x1FFFFFFu) | (2u << 30);// global_addr[56:32], type=2
  v8i g1;
  g1[0] = (int)((1u << 16)        // data_size = 2 bytes
              | (1u << 20)        // pad_enable
              | (5u << 22)        // pad_interval: 64 DWORDs (=256B row)
              | (3u << 25));      // pad_amount: 4 DWORDs (=16B)
  g1[1] = (int)(128u << 16);                               // tensor_dim0 = 128 (lo16)
  g1[2] = (int)((tensorRows & 0xFFFFu) << 16);             // tensor_dim0 hi=0 | tensor_dim1 lo16
  g1[3] = (int)(((tensorRows >> 16) & 0xFFFFu) | (128u << 16)); // tensor_dim1 hi | tile_dim0=128
  g1[4] = (int)rows;                                       // tile_dim1 = rows, tile_dim2 = 0
  g1[5] = 128;                                             // tensor_dim0_stride = 128
  g1[6] = 0;
  g1[7] = 0;
  v4i z4 = {0, 0, 0, 0};
  v8i z8 = {0, 0, 0, 0, 0, 0, 0, 0};
  __builtin_amdgcn_tensor_load_to_lds(g0, g1, z4, z4, z8, 0);
}

// ---------------- Phase 0: transpose-convert MI recurrent weights to bf16 [n][k]
__global__ void k_cvt_T(const float* Ui, const float* Uc, const float* Uf,
                        const float* Uo, const float* Wa, unsigned short* UT) {
  int j = blockIdx.x;
  const float* src;
  if (j < 10)       src = Ui + (size_t)j * H_ * H_;
  else if (j < 20)  src = Uc + (size_t)(j - 10) * H_ * H_;
  else if (j == 20) src = Uf;
  else if (j == 21) src = Uo;
  else              src = Wa;
  unsigned short* dst = UT + (size_t)j * H_ * H_;
  for (int idx = threadIdx.x; idx < H_ * H_; idx += blockDim.x) {
    int k = idx >> 7, n = idx & 127;
    dst[n * H_ + k] = f2bf(src[k * H_ + n]);
  }
}

// ---------------- Phase A: fused 10x LSTM, g = [x|h] @ [Wih;Whh]^T + b, weights LDS-resident
#define KA_LDK 264
#define KA_LDG 520
__global__ __launch_bounds__(256)
void k_lstm(const float* __restrict__ feat, const float* __restrict__ Wih,
            const float* __restrict__ Whh, const float* __restrict__ bih,
            const float* __restrict__ bhh, unsigned short* __restrict__ tilde) {
  extern __shared__ char smem[];
  unsigned short* wlds = (unsigned short*)smem;            // [512][264] bf16 ([n][k])
  unsigned short* hx   = wlds + 512 * KA_LDK;              // [16][264]  bf16 [x|h]
  float*          g    = (float*)(hx + 16 * KA_LDK);       // [16][520]  f32 gates
  float*          bsh  = g + 16 * KA_LDG;                  // [512]

  int tid = threadIdx.x;
  int s = blockIdx.x >> 3, m0 = (blockIdx.x & 7) * 16;
  const float* wih = Wih + (size_t)s * G4H * H_;
  const float* whh = Whh + (size_t)s * G4H * H_;

  for (int idx = tid; idx < G4H * 256; idx += 256) {
    int n = idx >> 8, k = idx & 255;
    float v = (k < H_) ? wih[n * H_ + k] : whh[n * H_ + (k - H_)];
    wlds[n * KA_LDK + k] = f2bf(v);
  }
  for (int idx = tid; idx < G4H; idx += 256)
    bsh[idx] = bih[s * G4H + idx] + bhh[s * G4H + idx];

  int m = tid >> 4, h0 = (tid & 15) * 8;
  float creg[8];
#pragma unroll
  for (int e = 0; e < 8; e++) creg[e] = 0.0f;
  {
    const float* xp = feat + (((size_t)s * T_ + 0) * B_ + (m0 + m)) * H_ + h0;
#pragma unroll
    for (int e = 0; e < 8; e++) {
      hx[m * KA_LDK + h0 + e]      = f2bf(xp[e]);
      hx[m * KA_LDK + H_ + h0 + e] = 0;
    }
  }
  __syncthreads();

  int wave = tid >> 5, lane = tid & 31;
  for (int t = 0; t < T_; t++) {
    v16bf afr[8];
#pragma unroll
    for (int kk = 0; kk < 8; kk++) afr[kk] = load_a(hx, KA_LDK, kk * 32, lane);
#pragma unroll
    for (int nt = 0; nt < 4; nt++) {
      int n0 = wave * 64 + nt * 16;
      float bv = bsh[n0 + (lane & 15)];
      v8f acc = {bv, bv, bv, bv, bv, bv, bv, bv};
#pragma unroll
      for (int kk = 0; kk < 8; kk++)
        acc = wmma_bf16(afr[kk], load_b(wlds, KA_LDK, n0, kk * 32, lane), acc);
      store_c(g, KA_LDG, n0, lane, acc);
    }
    __syncthreads();
    {
      const float* xnext = feat + (((size_t)s * T_ + (t + 1)) * B_ + (m0 + m)) * H_ + h0;
      if (t + 2 < T_)
        __builtin_prefetch(feat + (((size_t)s * T_ + (t + 2)) * B_ + (m0 + m)) * H_ + h0, 0, 1);
#pragma unroll
      for (int e = 0; e < 8; e++) {
        int h = h0 + e;
        float gi = g[m * KA_LDG + h];
        float gf = g[m * KA_LDG + H_ + h];
        float gg = g[m * KA_LDG + 2 * H_ + h];
        float go = g[m * KA_LDG + 3 * H_ + h];
        creg[e] = sigm(gf) * creg[e] + sigm(gi) * tanhf(gg);
        float hv = sigm(go) * tanhf(creg[e]);
        tilde[(((size_t)s * T_ + t) * B_ + (m0 + m)) * H_ + h] = f2bf(fmaxf(hv, 0.0f));
        hx[m * KA_LDK + H_ + h] = f2bf(hv);
        if (t + 1 < T_) hx[m * KA_LDK + h] = f2bf(xnext[e]);
      }
    }
    __syncthreads();
  }
}

// ---------------- Phase B: time-parallel projections xi/xc/xf/xo (22 GEMMs, M=32768)
#define KB_LDK 136
__global__ __launch_bounds__(256)
void k_proj(const unsigned short* __restrict__ tilde,
            const float* Wi, const float* Wc, const float* Wf, const float* Wo,
            const float* bi, const float* bc, const float* bfv, const float* bo,
            float* __restrict__ xi, float* __restrict__ xc,
            float* __restrict__ xf, float* __restrict__ xo) {
  extern __shared__ char smem[];
  unsigned short* aT  = (unsigned short*)smem;        // [128][136] bf16 activations (TDM-filled)
  unsigned short* bT  = aT + 128 * KB_LDK;            // [128][136] bf16 weights [n][k]
  float*          bsh = (float*)(bT + 128 * KB_LDK);  // [128]

  int job = blockIdx.x >> 8, mblk = blockIdx.x & 255, tid = threadIdx.x;
  int wave = tid >> 5, lane = tid & 31;
  int sIdx = (job < 10) ? job : (job < 20 ? job - 10 : 0);
  const unsigned short* asrc = tilde + (size_t)sIdx * T_ * B_ * H_ + (size_t)mblk * 128 * H_;
  const float *wsrc, *bsrc;
  if (job < 10)       { wsrc = Wi + (size_t)job * H_ * H_;        bsrc = bi + job * H_; }
  else if (job < 20)  { wsrc = Wc + (size_t)(job - 10) * H_ * H_; bsrc = bc + (job - 10) * H_; }
  else if (job == 20) { wsrc = Wf; bsrc = bfv; }
  else                { wsrc = Wo; bsrc = bo; }

  // Wave 0 kicks off the async TDM copy of the 128x128 bf16 activation tile into
  // padded LDS; all waves convert the weight tile meanwhile.
  if (wave == 0)
    tdm_load_tile_bf16_r128(asrc, aT, 128, (unsigned)(T_ * B_));

  for (int idx = tid; idx < 128 * H_; idx += 256) {
    int k = idx >> 7, n = idx & 127;
    bT[n * KB_LDK + k] = f2bf(wsrc[idx]);   // [n][k] = W[k][n]
  }
  for (int idx = tid; idx < 128; idx += 256) bsh[idx] = bsrc[idx];

  if (wave == 0) __builtin_amdgcn_s_wait_tensorcnt(0);
  __syncthreads();

  v16bf afr[4];
#pragma unroll
  for (int kk = 0; kk < 4; kk++) afr[kk] = load_a(aT + wave * 16 * KB_LDK, KB_LDK, kk * 32, lane);
  int mh = (lane >> 4) * 8, nlo = lane & 15;
#pragma unroll
  for (int nt = 0; nt < 8; nt++) {
    int n0 = nt * 16;
    float bv = bsh[n0 + nlo];
    v8f acc = {bv, bv, bv, bv, bv, bv, bv, bv};
#pragma unroll
    for (int kk = 0; kk < 4; kk++)
      acc = wmma_bf16(afr[kk], load_b(bT, KB_LDK, n0, kk * 32, lane), acc);
    int n = n0 + nlo;
#pragma unroll
    for (int r = 0; r < 8; r++) {
      int mrow = mblk * 128 + wave * 16 + mh + r;
      if (job < 20) {
        int t = mrow >> 7, b = mrow & 127;
        int sj = (job < 10) ? job : job - 10;
        float* dst = (job < 10) ? xi : xc;
        dst[(((size_t)t * S_ + sj) * B_ + b) * H_ + n] = acc[r];
      } else {
        float* dst = (job == 20) ? xf : xo;
        dst[(size_t)mrow * H_ + n] = acc[r];
      }
    }
  }
}

// ---------------- Phase C: MI-LSTM scan + attention + fused head
#define KC_LDK 136
__global__ __launch_bounds__(256)
void k_milstm(const unsigned short* __restrict__ UT, const float* __restrict__ xi,
              const float* __restrict__ xc, const float* __restrict__ xf,
              const float* __restrict__ xo, const float* __restrict__ headW,
              const float* __restrict__ headB, float* __restrict__ out) {
  extern __shared__ char smem[];
  unsigned short* hbf = (unsigned short*)smem;        // [16][136]
  unsigned short* cbf = hbf + 16 * KC_LDK;            // [16][136]
  float* g    = (float*)(cbf + 16 * KC_LDK);          // [23][16][136]
  float* red  = g + 23 * 16 * KC_LDK;                 // [10][16][16]
  float* uar  = red + 10 * 16 * 16;                   // [10][16]
  float* alp  = uar + 10 * 16;                        // [10][16]
  float* hred = alp + 10 * 16;                        // [16][16]

  int tid = threadIdx.x, m0 = blockIdx.x * 16;
  int m = tid >> 4, h0 = (tid & 15) * 8;
  int wave = tid >> 5, lane = tid & 31;
  float creg[8];
#pragma unroll
  for (int e = 0; e < 8; e++) creg[e] = 0.0f;
  for (int idx = tid; idx < 16 * KC_LDK; idx += 256) { hbf[idx] = 0; cbf[idx] = 0; }
  __syncthreads();

  float hW[8];
#pragma unroll
  for (int e = 0; e < 8; e++) hW[e] = headW[h0 + e];
  float hb = headB[0];

  for (int t = 0; t < T_; t++) {
    v16bf afrH[4], afrC[4];
#pragma unroll
    for (int kk = 0; kk < 4; kk++) {
      afrH[kk] = load_a(hbf, KC_LDK, kk * 32, lane);
      afrC[kk] = load_a(cbf, KC_LDK, kk * 32, lane);
    }
    for (int tile = wave; tile < 184; tile += 8) {      // 23 products x 8 ntiles
      int j = tile >> 3, nt = tile & 7, n0 = nt * 16;
      const unsigned short* bsrc = UT + (size_t)j * H_ * H_;
      v8f acc = {0, 0, 0, 0, 0, 0, 0, 0};
#pragma unroll
      for (int kk = 0; kk < 4; kk++)
        acc = wmma_bf16((j == 22) ? afrC[kk] : afrH[kk],
                        load_b(bsrc, H_, n0, kk * 32, lane), acc);
      store_c(g + j * 16 * KC_LDK, KC_LDK, n0, lane, acc);
    }
    __syncthreads();

    int brow = m0 + m;
    float fg[8], og[8], wv[8], part[10];
#pragma unroll
    for (int s = 0; s < 10; s++) part[s] = 0.0f;
#pragma unroll
    for (int e = 0; e < 8; e++) {
      int h = h0 + e;
      wv[e] = g[(22 * 16 + m) * KC_LDK + h];
      fg[e] = sigm(xf[((size_t)t * B_ + brow) * H_ + h] + g[(20 * 16 + m) * KC_LDK + h]);
      og[e] = sigm(xo[((size_t)t * B_ + brow) * H_ + h] + g[(21 * 16 + m) * KC_LDK + h]);
    }
    if (t + 1 < T_) {
      __builtin_prefetch(xi + (((size_t)(t + 1) * S_) * B_ + brow) * H_ + h0, 0, 1);
      __builtin_prefetch(xc + (((size_t)(t + 1) * S_) * B_ + brow) * H_ + h0, 0, 1);
    }
    for (int s = 0; s < 10; s++) {
      const float* xip = xi + (((size_t)t * S_ + s) * B_ + brow) * H_ + h0;
      const float* xcp = xc + (((size_t)t * S_ + s) * B_ + brow) * H_ + h0;
#pragma unroll
      for (int e = 0; e < 8; e++) {
        int h = h0 + e;
        float iv = sigm(xip[e] + g[(s * 16 + m) * KC_LDK + h]);
        float cv = tanhf(xcp[e] + g[((10 + s) * 16 + m) * KC_LDK + h]);
        float l = iv * cv;
        g[(s * 16 + m) * KC_LDK + h] = l;          // reuse slot for l
        part[s] += l * wv[e];
      }
    }
#pragma unroll
    for (int s = 0; s < 10; s++) red[(s * 16 + m) * 16 + (tid & 15)] = part[s];
    __syncthreads();
    if (tid < 160) {
      int s = tid >> 4, mm = tid & 15;
      float a = 0.0f;
#pragma unroll
      for (int i = 0; i < 16; i++) a += red[(s * 16 + mm) * 16 + i];
      uar[s * 16 + mm] = tanhf(a);
    }
    __syncthreads();
    if (tid < 16) {
      float mx = -1e30f;
      for (int s = 0; s < 10; s++) mx = fmaxf(mx, uar[s * 16 + tid]);
      float den = 0.0f, ex[10];
      for (int s = 0; s < 10; s++) { ex[s] = expf(uar[s * 16 + tid] - mx); den += ex[s]; }
      float inv = 1.0f / den;
      for (int s = 0; s < 10; s++) alp[s * 16 + tid] = ex[s] * inv;
    }
    __syncthreads();
    float hp = 0.0f;
#pragma unroll
    for (int e = 0; e < 8; e++) {
      int h = h0 + e;
      float lmix = 0.0f;
      for (int s = 0; s < 10; s++) lmix += alp[s * 16 + m] * g[(s * 16 + m) * KC_LDK + h];
      float cn = fg[e] * creg[e] + lmix;
      creg[e] = cn;
      float hv = og[e] * tanhf(cn);
      hbf[m * KC_LDK + h] = f2bf(hv);
      cbf[m * KC_LDK + h] = f2bf(cn);
      hp += fmaxf(hv, 0.0f) * hW[e];
    }
    hred[m * 16 + (tid & 15)] = hp;
    __syncthreads();
    if (tid < 16) {
      float a = 0.0f;
#pragma unroll
      for (int i = 0; i < 16; i++) a += hred[tid * 16 + i];
      out[(size_t)t * B_ + m0 + tid] = fmaxf(a + hb, 0.0f);
    }
  }
}

extern "C" void kernel_launch(void* const* d_in, const int* in_sizes, int n_in,
                              void* d_out, int out_size, void* d_ws, size_t ws_size,
                              hipStream_t stream) {
  const float* feat = (const float*)d_in[0];
  const float* Wih = (const float*)d_in[1];
  const float* Whh = (const float*)d_in[2];
  const float* bih = (const float*)d_in[3];
  const float* bhh = (const float*)d_in[4];
  const float* mWi = (const float*)d_in[5];
  const float* mUi = (const float*)d_in[6];
  const float* mbi = (const float*)d_in[7];
  const float* mWc = (const float*)d_in[8];
  const float* mUc = (const float*)d_in[9];
  const float* mbc = (const float*)d_in[10];
  const float* mWf = (const float*)d_in[11];
  const float* mUf = (const float*)d_in[12];
  const float* mbf = (const float*)d_in[13];
  const float* mWo = (const float*)d_in[14];
  const float* mUo = (const float*)d_in[15];
  const float* mbo = (const float*)d_in[16];
  const float* mWa = (const float*)d_in[17];
  const float* hW  = (const float*)d_in[18];
  const float* hb  = (const float*)d_in[19];
  float* out = (float*)d_out;

  char* ws = (char*)d_ws;
  size_t off = 0;
  auto carve = [&](size_t bytes) -> void* {
    void* p = ws + off;
    off = (off + bytes + 255) & ~(size_t)255;
    return p;
  };
  unsigned short* tilde = (unsigned short*)carve((size_t)S_ * T_ * B_ * H_ * 2);
  float* xi = (float*)carve((size_t)T_ * S_ * B_ * H_ * 4);
  float* xc = (float*)carve((size_t)T_ * S_ * B_ * H_ * 4);
  float* xf = (float*)carve((size_t)T_ * B_ * H_ * 4);
  float* xo = (float*)carve((size_t)T_ * B_ * H_ * 4);
  unsigned short* UT = (unsigned short*)carve((size_t)23 * H_ * H_ * 2);

  k_cvt_T<<<23, 256, 0, stream>>>(mUi, mUc, mUf, mUo, mWa, UT);

  size_t ldsA = (size_t)512 * KA_LDK * 2 + 16 * KA_LDK * 2 + 16 * KA_LDG * 4 + 512 * 4;
  k_lstm<<<S_ * (B_ / 16), 256, ldsA, stream>>>(feat, Wih, Whh, bih, bhh, tilde);

  size_t ldsB = (size_t)128 * KB_LDK * 2 * 2 + 128 * 4;
  k_proj<<<22 * 256, 256, ldsB, stream>>>(tilde, mWi, mWc, mWf, mWo,
                                          mbi, mbc, mbf, mbo, xi, xc, xf, xo);

  size_t ldsC = (size_t)16 * KC_LDK * 2 * 2 + (size_t)23 * 16 * KC_LDK * 4 +
                10 * 16 * 16 * 4 + 10 * 16 * 4 + 10 * 16 * 4 + 16 * 16 * 4;
  k_milstm<<<B_ / 16, 256, ldsC, stream>>>(UT, xi, xc, xf, xo, hW, hb, out);
}
